// EqualizedConv2dModulated_79267916415076
// MI455X (gfx1250) — matrix operand order repaired and checked by reading, compiled, not run on hardware
//
#include <hip/hip_runtime.h>
#include <hip/hip_bf16.h>

// CDNA5 / gfx1250: wave32, WMMA bf16 (16x16x32), f32 accumulate.

typedef __attribute__((ext_vector_type(16))) __bf16 v16bf;
typedef __attribute__((ext_vector_type(8)))  __bf16 v8bf;
typedef __attribute__((ext_vector_type(8)))  float  v8f;
typedef __attribute__((ext_vector_type(4)))  float  v4f;
typedef __attribute__((ext_vector_type(4)))  __bf16 v4bf;

#define IN_CH   512
#define OUT_CH  512
#define BATCH   16
#define HW      1024          // 32*32
#define KTOT    4608          // 9 * IN_CH, k = tap*512 + cin
#define WSCALE  0.0147313913f // 1/sqrt(512*9)
#define NSTEPS  144           // KTOT / 32

// ---------------------------------------------------------------------------
// Kernel 1: per-(b,cout) modulation + demodulation -> bf16 weights in d_ws
// layout: wm[b][cout][tap*512 + cin]
// ---------------------------------------------------------------------------
__global__ __launch_bounds__(256)
void eqconv_modulate_kernel(const float* __restrict__ s,
                            const float* __restrict__ w,
                            __bf16* __restrict__ wm) {
    const int co  = blockIdx.x;
    const int b   = blockIdx.y;
    const int tid = threadIdx.x;                 // 256 threads, 18 elems each
    const float* wrow = w + (size_t)co * KTOT;   // [cin][tap], tap fastest
    const float* srow = s + b * IN_CH;

    const int cin0 = tid * 2;                    // tid*18 = (tid*2)*9
    const float s0 = srow[cin0]     * WSCALE;
    const float s1 = srow[cin0 + 1] * WSCALE;

    float vals[18];
    float ss = 0.f;
#pragma unroll
    for (int i = 0; i < 18; ++i) {
        float v = (i < 9 ? s0 : s1) * wrow[tid * 18 + i];
        vals[i] = v;
        ss += v * v;
    }

    __shared__ float red[256];
    red[tid] = ss;
    __syncthreads();
    for (int off = 128; off > 0; off >>= 1) {
        if (tid < off) red[tid] += red[tid + off];
        __syncthreads();
    }
    const float inv = rsqrtf(red[0] + 1e-8f);

    __bf16* orow = wm + ((size_t)(b * OUT_CH + co)) * KTOT;
#pragma unroll
    for (int i = 0; i < 18; ++i) {
        const int cin = cin0 + (i / 9);
        const int tap = i % 9;
        orow[tap * IN_CH + cin] = (__bf16)(vals[i] * inv);
    }
}

// ---------------------------------------------------------------------------
// Kernel 2: fp32 x -> bf16 x (one pass; makes hot-loop gathers 2B/elem)
// ---------------------------------------------------------------------------
__global__ __launch_bounds__(256)
void eqconv_xcast_kernel(const float* __restrict__ x, __bf16* __restrict__ xb) {
    const int i = (blockIdx.x * 256 + threadIdx.x) * 4;
    v4f v = *(const v4f*)(x + i);
    v4bf o;
#pragma unroll
    for (int j = 0; j < 4; ++j) o[j] = (__bf16)v[j];
    *(v4bf*)(xb + i) = o;
}

// ---------------------------------------------------------------------------
// Kernel 3: implicit GEMM per sample: out[512 x 1024] = Wm[512 x 4608] * im2col(x)
// Block tile: M=128 x N=128, K-step=32. 8 waves as 4(M) x 2(N); each wave owns
// a 32x64 tile = 2x4 accumulators -> 8 WMMAs per K-step.
// Pipeline: issue global loads -> barrier -> WMMA from current LDS buffer ->
// wait loads + ds-store into next buffer (global latency hidden behind WMMA).
// LDS tiles stored as per-lane fragments in contiguous k order (k = 16*(lane>>4) + elem).
// ---------------------------------------------------------------------------
__global__ __launch_bounds__(256)
void eqconv_wmma_kernel(const __bf16* __restrict__ xbf,
                        const __bf16* __restrict__ wm,
                        float* __restrict__ out) {
    __shared__ __align__(32) __bf16 ldsA[2][128 * 32]; // 8 KB per buffer
    __shared__ __align__(32) __bf16 ldsB[2][32 * 128]; // 8 KB per buffer

    const int tid  = threadIdx.x;
    const int lane = tid & 31;
    const int wv   = tid >> 5;      // 0..7
    const int wM   = wv >> 1;       // 0..3  (32-row slice of 128 M-tile)
    const int wN   = wv & 1;        // 0..1  (64-col slice of 128 N-tile)

    const int b        = blockIdx.z;
    const int coutBase = blockIdx.y * 128;
    const int pBase    = blockIdx.x * 128;

    const __bf16* xb  = xbf + (size_t)b * IN_CH * HW;
    const __bf16* wmb = wm + ((size_t)(b * OUT_CH + coutBase)) * KTOT;

    v8f acc[2][4] = {};

    // A-tile loader: thread -> (row m, K-half). 128x32 bf16.
    const int am = tid >> 1;        // 0..127
    const int ag = tid & 1;         // k 0..15 or 16..31
    // B-tile loader: thread -> (col n, 16 K rows). 32x128 bf16.
    const int bn  = tid & 127;           // spatial column within N tile
    const int bk0 = (tid >> 7) << 4;     // 0 or 16

    const int p  = pBase + bn;
    const int py = p >> 5;
    const int px = p & 31;

    const __bf16 zbf = (__bf16)0.0f;

    v8bf  r_alo, r_ahi;       // staged A regs
    v16bf r_b;                // staged B regs

    auto issueLoads = [&](int ks) {
        const int tap  = ks >> 4;            // 16 K-steps per tap
        const int cinB = (ks & 15) << 5;

        // ---- A: 32 contiguous bf16 from modulated weights ----
        const __bf16* asrc = wmb + (size_t)am * KTOT + tap * IN_CH + cinB + ag * 16;
        r_alo = *(const v8bf*)asrc;          // k = ag*16 + 0..7
        r_ahi = *(const v8bf*)(asrc + 8);    // k = ag*16 + 8..15

        // ---- B: im2col column (16 cin rows), SAME-padding predication ----
        const int dy = tap / 3 - 1, dx = tap % 3 - 1;
        const int yy = py + dy, xx = px + dx;
        const bool valid = ((unsigned)yy < 32u) && ((unsigned)xx < 32u);
        const int  sp    = yy * 32 + xx;
        const __bf16* xcol = xb + (size_t)(cinB + bk0) * HW + sp;
        alignas(32) __bf16 t[16];
#pragma unroll
        for (int j = 0; j < 16; ++j)
            t[j] = valid ? xcol[(size_t)j * HW] : zbf;
        r_b = *(const v16bf*)t;
    };

    auto storeTile = [&](int buf) {
        // A fragment order: lane = (m&15) + 16*(k>>4), elem = k&15
        __bf16* abase = &ldsA[buf][(am >> 4) * 512];
        *(v8bf*)&abase[((am & 15) + 16 * ag) * 16    ] = r_alo;
        *(v8bf*)&abase[((am & 15) + 16 * ag) * 16 + 8] = r_ahi;
        // B fragment order: lane = (n&15) + 16*(k>=16), elem = k&15
        const int blane = (bn & 15) + (bk0 ? 16 : 0);
        *(v16bf*)&ldsB[buf][(bn >> 4) * 512 + blane * 16] = r_b;
    };

    // prologue: fill buffer 0
    issueLoads(0);
    storeTile(0);

#pragma unroll 1
    for (int ks = 0; ks < NSTEPS; ++ks) {
        const bool more = (ks + 1 < NSTEPS);
        if (more) issueLoads(ks + 1);        // global loads in flight...
        __syncthreads();                     // buffer ks&1 now complete

        const int buf = ks & 1;
        v16bf afrag[2], bfrag[4];
#pragma unroll
        for (int mi = 0; mi < 2; ++mi)
            afrag[mi] = *(const v16bf*)&ldsA[buf][(wM * 2 + mi) * 512 + lane * 16];
#pragma unroll
        for (int ni = 0; ni < 4; ++ni)
            bfrag[ni] = *(const v16bf*)&ldsB[buf][(wN * 4 + ni) * 512 + lane * 16];

#pragma unroll
        for (int mi = 0; mi < 2; ++mi)
#pragma unroll
            for (int ni = 0; ni < 4; ++ni)
                acc[mi][ni] = __builtin_amdgcn_wmma_f32_16x16x32_bf16(
                    false, afrag[mi], false, bfrag[ni],
                    (short)0, acc[mi][ni], false, false);

        if (more) storeTile((ks + 1) & 1);   // ...consumed after the WMMAs
    }

    // ---- epilogue: C layout VGPR j -> M = j + 8*(lane>=16), N = lane&15 ----
    const int mrow = 8 * (lane >> 4);
    const int ncol = lane & 15;
#pragma unroll
    for (int mi = 0; mi < 2; ++mi)
#pragma unroll
        for (int ni = 0; ni < 4; ++ni) {
            const int cout0 = coutBase + wM * 32 + mi * 16 + mrow;
            const int p0    = pBase    + wN * 64 + ni * 16 + ncol;
#pragma unroll
            for (int j = 0; j < 8; ++j)
                out[((size_t)(b * OUT_CH + cout0 + j)) * HW + p0] = acc[mi][ni][j];
        }
}

// ---------------------------------------------------------------------------
extern "C" void kernel_launch(void* const* d_in, const int* in_sizes, int n_in,
                              void* d_out, int out_size, void* d_ws, size_t ws_size,
                              hipStream_t stream) {
    (void)in_sizes; (void)n_in; (void)out_size; (void)ws_size;
    const float* x = (const float*)d_in[0];   // [16,512,32,32]
    const float* s = (const float*)d_in[1];   // [16,512]
    const float* w = (const float*)d_in[2];   // [512,512,3,3]
    float* out = (float*)d_out;               // [16,512,32,32]

    __bf16* wm  = (__bf16*)d_ws;                                  // 75.5 MB
    __bf16* xbf = (__bf16*)((char*)d_ws + (size_t)BATCH * OUT_CH * KTOT * 2);

    eqconv_modulate_kernel<<<dim3(OUT_CH, BATCH), 256, 0, stream>>>(s, w, wm);
    eqconv_xcast_kernel<<<dim3(BATCH * IN_CH * HW / 1024), 256, 0, stream>>>(x, xbf);
    eqconv_wmma_kernel<<<dim3(HW / 128, OUT_CH / 128, BATCH), 256, 0, stream>>>(xbf, wm, out);
}
